// SNNIFFunc_86706799772108
// MI455X (gfx1250) — matrix operand order repaired and checked by reading, not compiled, not run on hardware
//
#include <hip/hip_runtime.h>

// ---------------------------------------------------------------------------
// SNN integrate-and-fire recurrence on MI455X (gfx1250).
//   x1 = x @ Wx^T + bx                      (one GEMM)
//   63x: u += s @ Ws^T + bs + x1 ; s = (u>=1); u -= s; a += s
// Compute-bound (~275 GFLOP vs ~34 MB irreducible HBM traffic):
//   * f16 WMMA (V_WMMA_F32_16X16X32_F16), f32 accumulation
//   * persistent workgroups: u lives in the WMMA accumulators, spike counts
//     in VGPRs, s / (x1+bs) in LDS; only W^T slices stream from L2.
//   * W^T streams via global_load_async_to_lds_b128 (ASYNCcnt DMA) through a
//     TRIPLE-buffered LDS ring: per slice the wave waits asynccnt<=16 (newest
//     slice may stay in flight), so DMA overlaps the whole WMMA burst and
//     only 9 barriers are needed per time step.
//   * xstep (= x1 + bs) is stored in C-fragment order so the per-step reload
//     is 2x ds_load_b128 per tile instead of 8x ds_load_b32.
// ---------------------------------------------------------------------------

typedef _Float16 half_t;
typedef __attribute__((ext_vector_type(8)))  _Float16 v8h;
typedef __attribute__((ext_vector_type(16))) _Float16 v16h;
typedef __attribute__((ext_vector_type(8)))  float    v8f;

constexpr int D       = 512;              // feature dim
constexpr int MT      = 32;               // batch rows per workgroup
constexpr int KS      = 64;               // K depth staged per slice
constexpr int SLICE   = KS * D;           // halves per contiguous global slice
constexpr int NSLICE  = D / KS;           // 8 K-slices per GEMM
constexpr int NT      = 8;                // 16-wide N tiles per wave
constexpr int THREADS = 256;              // 8 waves (wave32)
constexpr int NWAVES  = THREADS / 32;

// Padded LDS row strides: 528 halves = 1056 B = 264 DWORDs, 264 % 64 = 8 ->
// consecutive rows rotate banks by 8 -> 32-lane b128 reads resolve in the
// minimal 4 phases.
constexpr int SST = D + 16;               // s / x tile (halves)
constexpr int WST = D + 16;               // W slice rows (halves)
constexpr int WBUF = KS * WST;            // halves per W ring buffer entry

constexpr int CH_W = (KS * D / 8) / THREADS;   // 16B chunks/thread per W slice = 16
constexpr int CH_S = (MT * D / 8) / THREADS;   // 16B chunks/thread per x tile  = 8

constexpr size_t SMEM_S = (size_t)MT * SST * 2;          //  33792 B
constexpr size_t SMEM_W = (size_t)3 * WBUF * 2;          // 202752 B (ring of 3)
constexpr size_t SMEM_X = (size_t)NWAVES * NT * 32 * 8 * 4; // 65536 B (frag order)
constexpr size_t SMEM_TOTAL = SMEM_S + SMEM_W + SMEM_X;  // 302080 B < 320 KB

// --- CDNA5 async global->LDS DMA (ASYNCcnt-tracked, no VGPR data path) ----
// Stage a contiguous [ROWSxD] f16 tile from global into row-padded LDS.
// Lanes issue consecutive 16B chunks -> fully coalesced 512B burst per wave.
template <int NCH>
__device__ __forceinline__ void stage_async(const half_t* __restrict__ g,
                                            half_t* __restrict__ l, int tid) {
    #pragma unroll
    for (int i = 0; i < NCH; ++i) {
        int c  = tid + THREADS * i;       // 16B chunk id
        int kl = c >> 6;                  // row (64 chunks per 512-half row)
        int nc = (c & 63) << 3;           // halves within row
        unsigned lds = (unsigned)(unsigned long long)(const void*)(l + kl * WST + nc);
        unsigned long long ga = (unsigned long long)(const void*)(g + (size_t)c * 8);
        asm volatile("global_load_async_to_lds_b128 %0, %1, off"
                     :: "v"(lds), "v"(ga) : "memory");
    }
}

// Async loads complete in order: allowing CH_W outstanding means "everything
// except the most recently issued slice has landed".
__device__ __forceinline__ void wait_async_tail() {
    asm volatile("s_wait_asynccnt 16" ::: "memory");
}
__device__ __forceinline__ void wait_async0() {
    asm volatile("s_wait_asynccnt 0x0" ::: "memory");
}

// A fragment (16x32 f16): lanes 0-15 hold row M, halves = K {0..7,16..23};
// lanes 16-31 same rows, halves = K {8..15,24..31}. kq = K-chunk-of-32 index.
__device__ __forceinline__ v16h load_afrag(const half_t* __restrict__ a_lds,
                                           int mBase, int l15, int hi, int kq) {
    const half_t* ap = a_lds + (mBase + l15) * SST + kq * 32 + hi * 8;
    v8h alo = *(const v8h*)ap;
    v8h ahi = *(const v8h*)(ap + 16);
    return __builtin_shufflevector(alo, ahi,
        0, 1, 2, 3, 4, 5, 6, 7, 8, 9, 10, 11, 12, 13, 14, 15);
}

// One full K=512 GEMM pass: acc[nt] += A(a_lds tile) x B(wT), chained on the
// incoming acc (which carries the membrane potential u).
// Precondition: slice 0 DMA already ISSUED by the caller (may be in flight).
// Postcondition: no trailing barrier -- caller must barrier before touching
// a_lds; the W ring slot for the next pass's slice 0 (buf0) is already safe.
__device__ __forceinline__ void gemm_pass(const half_t* __restrict__ wT,
                                          half_t* __restrict__ wbuf,
                                          const half_t* __restrict__ a_lds,
                                          v8f acc[NT], int mBase, int n0,
                                          int tid, int lane) {
    const int l15 = lane & 15;
    const int hi  = (lane >> 4) & 1;

    stage_async<CH_W>(wT + SLICE, wbuf + WBUF, tid);   // slice 1 -> ring[1]

    #pragma unroll 1
    for (int kc = 0; kc < NSLICE; ++kc) {
        // Slice kc resident; newest slice may still be in flight.
        if (kc + 1 < NSLICE) wait_async_tail(); else wait_async0();
        __syncthreads();   // also publishes caller's s_lds writes at kc==0

        const half_t* cur = wbuf + (kc % 3) * WBUF;
        v16h a0 = load_afrag(a_lds, mBase, l15, hi, 2 * kc);
        v16h a1 = load_afrag(a_lds, mBase, l15, hi, 2 * kc + 1);

        // B fragments (32x16 f16): lane column N = l15, halves = K {0..7,16..23} (lanes 0-15) / {8..15,24..31} (lanes 16-31).
        const half_t* bb0 = cur + (hi * 8) * WST + n0 + l15;
        const half_t* bb1 = cur + (32 + hi * 8) * WST + n0 + l15;
        #pragma unroll
        for (int nt = 0; nt < NT; ++nt) {
            v16h b;
            #pragma unroll
            for (int s = 0; s < 16; ++s) b[s] = bb0[nt * 16 + (s + (s & 8)) * WST];
            acc[nt] = __builtin_amdgcn_wmma_f32_16x16x32_f16(
                false, a0, false, b, (short)0, acc[nt], false, false);
        }
        #pragma unroll
        for (int nt = 0; nt < NT; ++nt) {
            v16h b;
            #pragma unroll
            for (int s = 0; s < 16; ++s) b[s] = bb1[nt * 16 + (s + (s & 8)) * WST];
            acc[nt] = __builtin_amdgcn_wmma_f32_16x16x32_f16(
                false, a1, false, b, (short)0, acc[nt], false, false);
        }

        // Ring slot (kc+2)%3 was last read at iteration kc-1; every wave has
        // passed this iteration's barrier since then -> safe to overwrite.
        if (kc + 2 < NSLICE)
            stage_async<CH_W>(wT + (size_t)(kc + 2) * SLICE,
                              wbuf + ((kc + 2) % 3) * WBUF, tid);
    }
}

__global__ __launch_bounds__(THREADS) void snn_if_kernel(
    const half_t* __restrict__ xh,    // [B][D] f16 (pre-converted)
    const half_t* __restrict__ WxT,   // [D][D] f16, K-major
    const half_t* __restrict__ WsT,   // [D][D] f16, K-major
    const float* __restrict__ bs,
    const float* __restrict__ bx,
    const int* __restrict__ tstep,
    float* __restrict__ out) {
    extern __shared__ char smem[];
    half_t* s_lds = (half_t*)smem;                     // spike tile (A op)
    half_t* w_lds = (half_t*)(smem + SMEM_S);          // W slice ring (x3)
    float*  xstep = (float*)(smem + SMEM_S + SMEM_W);  // x1+bs, fragment order

    const int tid   = threadIdx.x;
    const int lane  = tid & 31;
    const int wave  = tid >> 5;
    const int l15   = lane & 15;
    const int hi    = (lane >> 4) & 1;
    const int mBase = (wave >> 2) * 16;        // 0 / 16
    const int n0    = (wave & 3) * (NT * 16);  // 0 / 128 / 256 / 384
    const int rowBase = blockIdx.x * MT;
    const int T = *tstep;

    // Per-lane base of this wave's xstep fragments: [(wave,nt)][lane][r].
    float* xfrag = xstep + ((size_t)wave * NT * 256) + (size_t)lane * 8;

    float bsr[NT], bxr[NT];
    #pragma unroll
    for (int nt = 0; nt < NT; ++nt) {
        int n = n0 + nt * 16 + l15;
        bsr[nt] = bs[n];
        bxr[nt] = bx[n];
    }

    // DMA the x tile (A operand of the Wx GEMM) and Wx slice 0 into LDS.
    // The first pass's wait+barrier covers their completion.
    stage_async<CH_S>(xh + (size_t)rowBase * D, s_lds, tid);
    stage_async<CH_W>(WxT, w_lds, tid);

    v8f acc[NT];
    #pragma unroll
    for (int nt = 0; nt < NT; ++nt) { v8f z = {}; acc[nt] = z; }

    // acc = x @ Wx^T
    gemm_pass(WxT, w_lds, s_lds, acc, mBase, n0, tid, lane);

    // Prefetch Ws slice 0 (ring[0] last read two barriers ago) and fence the
    // pass's s_lds reads before the epilogue overwrites it with spikes.
    stage_async<CH_W>(WsT, w_lds, tid);
    __syncthreads();

    // t = 0: u = x1; spike; stash xstep = x1 + bs; spikes overwrite s_lds.
    // C/D layout: VGPR r <-> M = r (lanes 0-15) / r+8 (lanes 16-31), N = l15.
    float cnt[NT][8];
    #pragma unroll
    for (int nt = 0; nt < NT; ++nt) {
        const int n = n0 + nt * 16 + l15;
        float* xw = xfrag + nt * 256;          // 8 contiguous floats per lane
        #pragma unroll
        for (int r = 0; r < 8; ++r) {
            const int m = mBase + r + hi * 8;
            float x1 = acc[nt][r] + bxr[nt];
            xw[r] = x1 + bsr[nt];
            float sp = (x1 >= 1.0f) ? 1.0f : 0.0f;
            cnt[nt][r] = sp;
            acc[nt][r] = x1 - sp;                 // u after soft reset
            s_lds[m * SST + n] = (half_t)sp;      // spikes are exact in f16
        }
    }
    // No barrier here: the next pass's first wait+barrier publishes spikes.

    // Recurrent time loop: u (in acc) never leaves registers.
    #pragma unroll 1
    for (int t = 1; t < T; ++t) {
        gemm_pass(WsT, w_lds, s_lds, acc, mBase, n0, tid, lane); // u += s@Ws^T

        // Hoist next step's slice-0 DMA over the epilogue, then fence the
        // pass's s_lds reads before rewriting spikes.
        if (t + 1 < T) stage_async<CH_W>(WsT, w_lds, tid);
        __syncthreads();

        #pragma unroll
        for (int nt = 0; nt < NT; ++nt) {
            const int n = n0 + nt * 16 + l15;
            v8f xv = *(const v8f*)(xfrag + nt * 256);   // 2x ds_load_b128
            #pragma unroll
            for (int r = 0; r < 8; ++r) {
                const int m = mBase + r + hi * 8;
                float u  = acc[nt][r] + xv[r];
                float sp = (u >= 1.0f) ? 1.0f : 0.0f;
                cnt[nt][r] += sp;
                acc[nt][r] = u - sp;
                s_lds[m * SST + n] = (half_t)sp;
            }
        }
    }

    const float scale = 1.0f / (float)T;
    #pragma unroll
    for (int nt = 0; nt < NT; ++nt) {
        const int n = n0 + nt * 16 + l15;
        #pragma unroll
        for (int r = 0; r < 8; ++r) {
            const int row = rowBase + mBase + r + hi * 8;
            out[(size_t)row * D + n] = cnt[nt][r] * scale;
        }
    }
}

// ---- prep kernels: f32->f16 convert, and transpose weights to K-major ----
__global__ void cvt_f32_to_f16(const float* __restrict__ in,
                               half_t* __restrict__ outp, int n) {
    int i = blockIdx.x * blockDim.x + threadIdx.x;
    if (i < n) outp[i] = (half_t)in[i];
}

__global__ void transpose_to_f16(const float* __restrict__ w,
                                 half_t* __restrict__ wt) {
    // wt[k][n] = w[n][k]  (so B element (k,n) is contiguous in n)
    int idx = blockIdx.x * blockDim.x + threadIdx.x;
    int k = idx / D;
    int n = idx % D;
    wt[idx] = (half_t)w[n * D + k];
}

extern "C" void kernel_launch(void* const* d_in, const int* in_sizes, int n_in,
                              void* d_out, int out_size, void* d_ws, size_t ws_size,
                              hipStream_t stream) {
    const float* x  = (const float*)d_in[0];
    const float* Ws = (const float*)d_in[1];
    const float* bs = (const float*)d_in[2];
    const float* Wx = (const float*)d_in[3];
    const float* bx = (const float*)d_in[4];
    const int*   tp = (const int*)d_in[5];
    float* out = (float*)d_out;

    const int BD = in_sizes[0];          // B * D
    const int B  = BD / D;

    // workspace layout (needs BD*2 + 2*D*D*2 bytes ~= 9 MB)
    half_t* xh  = (half_t*)d_ws;
    half_t* WsT = (half_t*)((char*)d_ws + (size_t)BD * 2);
    half_t* WxT = WsT + (size_t)D * D;

    cvt_f32_to_f16<<<BD / 256, 256, 0, stream>>>(x, xh, BD);
    transpose_to_f16<<<(D * D) / 256, 256, 0, stream>>>(Ws, WsT);
    transpose_to_f16<<<(D * D) / 256, 256, 0, stream>>>(Wx, WxT);

    snn_if_kernel<<<dim3(B / MT), dim3(THREADS), SMEM_TOTAL, stream>>>(
        xh, WxT, WsT, bs, bx, tp, out);
}
